// SAGE_63651415327135
// MI455X (gfx1250) — matrix operand order repaired and checked by reading, not compiled, not run on hardware
//
#include <hip/hip_runtime.h>

typedef _Float16 v16h __attribute__((ext_vector_type(16)));
typedef _Float16 v8h  __attribute__((ext_vector_type(8)));
typedef float    v8f  __attribute__((ext_vector_type(8)));

#define N_NODES 50000
#define N_EDGES 800000
#define F_IN  128
#define F_HID 128
#define F_OUT 64

// ---------------- utility kernels ----------------

__global__ void zero_f32(float* __restrict__ p, int n) {
  for (int i = blockIdx.x * blockDim.x + threadIdx.x; i < n;
       i += gridDim.x * blockDim.x)
    p[i] = 0.0f;
}

__global__ void deg_count(const int* __restrict__ dst, float* __restrict__ deg) {
  for (int e = blockIdx.x * blockDim.x + threadIdx.x; e < N_EDGES;
       e += gridDim.x * blockDim.x)
    unsafeAtomicAdd(&deg[dst[e]], 1.0f);
}

__global__ void inv_deg_k(const float* __restrict__ deg, float* __restrict__ inv) {
  int i = blockIdx.x * blockDim.x + threadIdx.x;
  if (i < N_NODES) inv[i] = 1.0f / fmaxf(deg[i], 1.0f);
}

// one wave32 per edge: 128-float feature row = 32 lanes x float4,
// hardware f32 atomics scatter-add into the destination row (L2 resident).
__global__ __launch_bounds__(256) void edge_agg(
    const float* __restrict__ feat, const int* __restrict__ src,
    const int* __restrict__ dst, float* __restrict__ agg) {
  int e = blockIdx.x * 8 + (threadIdx.x >> 5);
  if (e >= N_EDGES) return;
  int lane = threadIdx.x & 31;
  int s = src[e], d = dst[e];
  const float4 v = *(const float4*)(feat + (size_t)s * 128 + lane * 4);
  float* p = agg + (size_t)d * 128 + lane * 4;
  unsafeAtomicAdd(p + 0, v.x);
  unsafeAtomicAdd(p + 1, v.y);
  unsafeAtomicAdd(p + 2, v.z);
  unsafeAtomicAdd(p + 3, v.w);
}

__global__ void cvt_f16(const float* __restrict__ in, _Float16* __restrict__ out, int n) {
  int i = blockIdx.x * blockDim.x + threadIdx.x;
  if (i < n) out[i] = (_Float16)in[i];
}

// agg (f32, per-node rows of `feat` entries) * inv_deg -> f16
__global__ void scale_cvt(const float* __restrict__ agg, const float* __restrict__ inv,
                          _Float16* __restrict__ out, int n, int feat) {
  int i = blockIdx.x * blockDim.x + threadIdx.x;
  if (i < n) out[i] = (_Float16)(agg[i] * inv[i / feat]);
}

// w[k][n] (row-major K x N, f32) -> wt[n][k] (f16), i.e. contiguous-K per output col
__global__ void transpose_w(const float* __restrict__ w, _Float16* __restrict__ wt,
                            int K, int N) {
  int i = blockIdx.x * blockDim.x + threadIdx.x;
  if (i < K * N) {
    int k = i / N, n = i % N;
    wt[(size_t)n * K + k] = (_Float16)w[i];
  }
}

// ---------------- WMMA fragment helpers ----------------
// A 16x32 f16 fragment (CDNA5 layout): lane m = lane&15 holds
//   halves 0..7  = K[kk + 8*(lane>>4)     .. +7]
//   halves 8..15 = K[kk + 8*(lane>>4) + 16 .. +23]
__device__ __forceinline__ v16h load_a_frag(const _Float16* __restrict__ base,
                                            int row, int ldk, int kk, int lane) {
  const _Float16* p = base + (size_t)row * ldk + kk + ((lane >> 4) << 3);
  v8h lo = *(const v8h*)(p);
  v8h hi = *(const v8h*)(p + 16);
  return __builtin_shufflevector(lo, hi, 0, 1, 2, 3, 4, 5, 6, 7,
                                 8, 9, 10, 11, 12, 13, 14, 15);
}

// B 32x16 f16 fragment from pre-transposed weights wt[n][k]:
// lane n = lane&15 holds halves 0..7 = K[kk + 8*(lane>>4) .. +7], 8..15 = K[kk + 8*(lane>>4) + 16 .. +23].
__device__ __forceinline__ v16h load_b_frag(const _Float16* __restrict__ wt,
                                            int col, int ldk, int kk, int lane) {
  const _Float16* p = wt + (size_t)col * ldk + kk + ((lane >> 4) << 3);
  v8h lo = *(const v8h*)(p);
  v8h hi = *(const v8h*)(p + 16);
  return __builtin_shufflevector(lo, hi, 0, 1, 2, 3, 4, 5, 6, 7,
                                 8, 9, 10, 11, 12, 13, 14, 15);
}

// ---------------- layer 1: h = relu(x@Ws1 + agg1@Wn1 + b1) ----------------
// 8 waves/block, wave w owns N-tile [16w,16w+16); block owns rows [16*bid, +16)
__global__ __launch_bounds__(256) void sage_gemm1(
    const _Float16* __restrict__ xh, const _Float16* __restrict__ aggh,
    const _Float16* __restrict__ wst, const _Float16* __restrict__ wnt,
    const float* __restrict__ bias, float* __restrict__ h,
    _Float16* __restrict__ hh) {
  const int lane = threadIdx.x & 31;
  const int wv   = threadIdx.x >> 5;
  const int m0   = blockIdx.x * 16;
  const int n0   = wv * 16;
  const int mA   = m0 + (lane & 15);
  const int nB   = n0 + (lane & 15);

  v8f c = {};
#pragma unroll
  for (int kk = 0; kk < F_IN; kk += 32) {
    v16h a  = load_a_frag(xh,   mA, F_IN, kk, lane);
    v16h b  = load_b_frag(wst,  nB, F_IN, kk, lane);
    c = __builtin_amdgcn_wmma_f32_16x16x32_f16(false, a, false, b, (short)0, c,
                                               false, false);
    v16h g  = load_a_frag(aggh, mA, F_IN, kk, lane);
    v16h bn = load_b_frag(wnt,  nB, F_IN, kk, lane);
    c = __builtin_amdgcn_wmma_f32_16x16x32_f16(false, g, false, bn, (short)0, c,
                                               false, false);
  }
  const float bs = bias[nB];
  const int mC = m0 + ((lane >> 4) << 3);
#pragma unroll
  for (int r = 0; r < 8; ++r) {
    float v = c[r] + bs;
    v = v > 0.0f ? v : 0.0f;
    h [(size_t)(mC + r) * F_HID + nB] = v;
    hh[(size_t)(mC + r) * F_HID + nB] = (_Float16)v;
  }
}

// ---------------- layer 2: out = h@Ws2 + agg2@Wn2 + b2 ----------------
// 4 waves/block (N=64)
__global__ __launch_bounds__(128) void sage_gemm2(
    const _Float16* __restrict__ hh, const _Float16* __restrict__ aggh,
    const _Float16* __restrict__ wst, const _Float16* __restrict__ wnt,
    const float* __restrict__ bias, float* __restrict__ out) {
  const int lane = threadIdx.x & 31;
  const int wv   = threadIdx.x >> 5;
  const int m0   = blockIdx.x * 16;
  const int n0   = wv * 16;
  const int mA   = m0 + (lane & 15);
  const int nB   = n0 + (lane & 15);

  v8f c = {};
#pragma unroll
  for (int kk = 0; kk < F_HID; kk += 32) {
    v16h a  = load_a_frag(hh,   mA, F_HID, kk, lane);
    v16h b  = load_b_frag(wst,  nB, F_HID, kk, lane);
    c = __builtin_amdgcn_wmma_f32_16x16x32_f16(false, a, false, b, (short)0, c,
                                               false, false);
    v16h g  = load_a_frag(aggh, mA, F_HID, kk, lane);
    v16h bn = load_b_frag(wnt,  nB, F_HID, kk, lane);
    c = __builtin_amdgcn_wmma_f32_16x16x32_f16(false, g, false, bn, (short)0, c,
                                               false, false);
  }
  const float bs = bias[nB];
  const int mC = m0 + ((lane >> 4) << 3);
#pragma unroll
  for (int r = 0; r < 8; ++r)
    out[(size_t)(mC + r) * F_OUT + nB] = c[r] + bs;
}

// ---------------- host launch ----------------

extern "C" void kernel_launch(void* const* d_in, const int* in_sizes, int n_in,
                              void* d_out, int out_size, void* d_ws, size_t ws_size,
                              hipStream_t stream) {
  const float* x   = (const float*)d_in[0];
  const float* ws1 = (const float*)d_in[1];
  const float* wn1 = (const float*)d_in[2];
  const float* b1  = (const float*)d_in[3];
  const float* ws2 = (const float*)d_in[4];
  const float* wn2 = (const float*)d_in[5];
  const float* b2  = (const float*)d_in[6];
  const int*   src = (const int*)d_in[7];
  const int*   dst = (const int*)d_in[8];
  float* out = (float*)d_out;

  // carve workspace (256B-aligned slices)
  char* base = (char*)d_ws;
  size_t off = 0;
  auto carve = [&](size_t bytes) -> void* {
    void* p = base + off;
    off += (bytes + 255) & ~(size_t)255;
    return p;
  };
  float*    deg  = (float*)   carve(sizeof(float) * N_NODES);
  float*    inv  = (float*)   carve(sizeof(float) * N_NODES);
  float*    agg  = (float*)   carve(sizeof(float) * (size_t)N_NODES * 128);
  float*    hf   = (float*)   carve(sizeof(float) * (size_t)N_NODES * F_HID);
  _Float16* xh   = (_Float16*)carve(2 * (size_t)N_NODES * F_IN);
  _Float16* aggh = (_Float16*)carve(2 * (size_t)N_NODES * 128);
  _Float16* hh   = (_Float16*)carve(2 * (size_t)N_NODES * F_HID);
  _Float16* ws1t = (_Float16*)carve(2 * F_IN * F_HID);
  _Float16* wn1t = (_Float16*)carve(2 * F_IN * F_HID);
  _Float16* ws2t = (_Float16*)carve(2 * F_HID * F_OUT);
  _Float16* wn2t = (_Float16*)carve(2 * F_HID * F_OUT);

  const int nfeat = N_NODES * 128;          // 6.4M
  const int cvt_blocks = (nfeat + 255) / 256;

  // degree + inv_deg
  zero_f32<<<(N_NODES + 255) / 256, 256, 0, stream>>>(deg, N_NODES);
  zero_f32<<<4096, 256, 0, stream>>>(agg, nfeat);
  deg_count<<<2048, 256, 0, stream>>>(dst, deg);
  inv_deg_k<<<(N_NODES + 255) / 256, 256, 0, stream>>>(deg, inv);

  // layer-1 aggregation (x) + conversions
  edge_agg<<<N_EDGES / 8, 256, 0, stream>>>(x, src, dst, agg);
  cvt_f16<<<cvt_blocks, 256, 0, stream>>>(x, xh, nfeat);
  scale_cvt<<<cvt_blocks, 256, 0, stream>>>(agg, inv, aggh, nfeat, 128);

  // weight transposes (tiny)
  transpose_w<<<(F_IN * F_HID + 255) / 256, 256, 0, stream>>>(ws1, ws1t, F_IN, F_HID);
  transpose_w<<<(F_IN * F_HID + 255) / 256, 256, 0, stream>>>(wn1, wn1t, F_IN, F_HID);
  transpose_w<<<(F_HID * F_OUT + 255) / 256, 256, 0, stream>>>(ws2, ws2t, F_HID, F_OUT);
  transpose_w<<<(F_HID * F_OUT + 255) / 256, 256, 0, stream>>>(wn2, wn2t, F_HID, F_OUT);

  // layer 1 GEMM (WMMA): 50000/16 = 3125 row tiles, 8 N-tiles per block
  sage_gemm1<<<N_NODES / 16, 256, 0, stream>>>(xh, aggh, ws1t, wn1t, b1, hf, hh);

  // layer-2 aggregation (h)
  zero_f32<<<4096, 256, 0, stream>>>(agg, nfeat);
  edge_agg<<<N_EDGES / 8, 256, 0, stream>>>(hf, src, dst, agg);
  scale_cvt<<<cvt_blocks, 256, 0, stream>>>(agg, inv, aggh, nfeat, 128);

  // layer 2 GEMM (WMMA): 4 N-tiles per block
  sage_gemm2<<<N_NODES / 16, 128, 0, stream>>>(hh, aggh, ws2t, wn2t, b2, out);
}
